// CrossModalAttention_9002251452589
// MI455X (gfx1250) — matrix operand order repaired and checked by reading, not compiled, not run on hardware
//
#include <hip/hip_runtime.h>
#include <hip/hip_bf16.h>
#include <math.h>

// ---------------- CDNA5 WMMA types ----------------
typedef __attribute__((ext_vector_type(16))) __bf16 v16bf;
typedef __attribute__((ext_vector_type(8)))  float  v8f;

#define D_MODEL 256
#define SEQ_L   2048
#define N_B     4
#define N_H     8
#define HEAD_D  32

// Fragment loader: 16 bytes at p, 16 bytes at p+gapElems (gap in bf16 elements).
// A-fragment (16x32, row-major source): gap = 16.  B-fragment (32x16, [n][k]
// row-major source): gap = 16 as well.
__device__ __forceinline__ v16bf load_frag(const __bf16* p, int gapElems) {
    union { v16bf v; float4 f[2]; } u;
    u.f[0] = *(const float4*)(p);
    u.f[1] = *(const float4*)(p + gapElems);
    return u.v;
}

__device__ __forceinline__ v8f wmma_bf16(v16bf a, v16bf b, v8f c) {
    return __builtin_amdgcn_wmma_f32_16x16x32_bf16(false, a, false, b,
                                                   (short)0, c, false, false);
}

// ---- CDNA5 async global->LDS (per-lane 16B, tracked by ASYNCcnt) ----
// LDS address operand is the wave-relative LDS byte address = low 32 bits of
// the generic pointer to a __shared__ object.
__device__ __forceinline__ void async_ld16(const void* lds_dst, const void* gsrc) {
    unsigned lds = (unsigned)(unsigned long long)lds_dst;
    unsigned long long g = (unsigned long long)gsrc;
    asm volatile("global_load_async_to_lds_b128 %0, %1, off"
                 :: "v"(lds), "v"(g) : "memory");
}
#define S_WAIT_ASYNCCNT(N) asm volatile("s_wait_asynccnt %0" :: "n"(N) : "memory")

// ---------------- Kernel 0: weights f32 -> bf16 ----------------
__global__ void __launch_bounds__(256) k_convert_w(
        const float* Wq, const float* Wk, const float* Wv, const float* Wo,
        __bf16* Wbf) {
    const int total = 4 * D_MODEL * D_MODEL;
    int i = blockIdx.x * blockDim.x + threadIdx.x;
    for (int j = i * 4; j < total; j += gridDim.x * blockDim.x * 4) {
        int m = j >> 16;                       // 65536 elems per matrix
        const float* src = (m == 0) ? Wq : (m == 1) ? Wk : (m == 2) ? Wv : Wo;
        float4 f = *(const float4*)(src + (j & 0xFFFF));
        __bf16* o = Wbf + j;
        o[0] = (__bf16)f.x; o[1] = (__bf16)f.y;
        o[2] = (__bf16)f.z; o[3] = (__bf16)f.w;
    }
}

// ---------------- Kernel 1: fused QKV projection ----------------
// 64 tokens per block, 256 threads = 8 wave32.  x-tile staged in LDS as bf16.
// Q,K stored row-major [B*L][256]; V stored transposed [B][256][L] so the
// attention kernel gets contiguous B-fragments for P*V.
__global__ void __launch_bounds__(256) k_qkv(
        const float* __restrict__ q_in, const float* __restrict__ k_in,
        const float* __restrict__ v_in, const __bf16* __restrict__ Wbf,
        const float* __restrict__ bq, const float* __restrict__ bk,
        const float* __restrict__ bv,
        __bf16* __restrict__ Qbf, __bf16* __restrict__ Kbf,
        __bf16* __restrict__ Vt) {
    __shared__ __bf16 xs[64][264];             // +8 pad, row stride 528B (16B mult)

    const int t0   = blockIdx.x * 64;          // global token base
    const int b    = t0 / SEQ_L;
    const int tl0  = t0 - b * SEQ_L;
    const int tid  = threadIdx.x;
    const int w    = tid >> 5, lane = tid & 31;
    const int half = lane >> 4, ln = lane & 15;

    for (int s = 0; s < 3; ++s) {
        const float* src  = (s == 0) ? q_in : (s == 1) ? k_in : v_in;
        const float* bias = (s == 0) ? bq   : (s == 1) ? bk   : bv;
        __syncthreads();
        // stage: 64x256 f32 -> bf16 LDS
        const float* xrow = src + (size_t)t0 * D_MODEL;
        for (int idx = tid * 4; idx < 64 * D_MODEL; idx += 256 * 4) {
            float4 f = *(const float4*)(xrow + idx);
            int r = idx >> 8, c = idx & 255;
            xs[r][c + 0] = (__bf16)f.x; xs[r][c + 1] = (__bf16)f.y;
            xs[r][c + 2] = (__bf16)f.z; xs[r][c + 3] = (__bf16)f.w;
        }
        __syncthreads();

        const __bf16* W = Wbf + s * D_MODEL * D_MODEL;  // [n][k] row-major
        for (int t = w; t < 64; t += 8) {               // 4 mtiles x 16 ntiles
            int mt = t >> 4, nt = t & 15;
            v8f acc = {};
            #pragma unroll
            for (int kk = 0; kk < D_MODEL; kk += 32) {
                v16bf a = load_frag(&xs[mt * 16 + ln][kk + 8 * half], 16);
                v16bf bm = load_frag(W + (size_t)(nt * 16 + ln) * D_MODEL
                                       + kk + 8 * half, 16);
                acc = wmma_bf16(a, bm, acc);
            }
            int n = nt * 16 + ln;
            float bb = bias[n];
            if (s < 2) {
                __bf16* dst = (s == 0) ? Qbf : Kbf;
                #pragma unroll
                for (int r = 0; r < 8; ++r) {
                    int m = mt * 16 + 8 * half + r;
                    dst[(size_t)(t0 + m) * D_MODEL + n] = (__bf16)(acc[r] + bb);
                }
            } else {                                    // V: transposed store
                union { __bf16 h[8]; float4 f; } pk;
                #pragma unroll
                for (int r = 0; r < 8; ++r) pk.h[r] = (__bf16)(acc[r] + bb);
                size_t o = (size_t)b * D_MODEL * SEQ_L + (size_t)n * SEQ_L
                         + tl0 + mt * 16 + 8 * half;    // 8 contiguous tokens
                *(float4*)(Vt + o) = pk.f;
            }
        }
    }
}

// ---------------- Kernel 2: flash attention ----------------
// grid = (L/64, B*H), block = 128 (4 waves).  Each wave owns 16 q-rows.
// K/V 32-key slices are shared by all 4 waves: staged to LDS with
// double-buffered GLOBAL_LOAD_ASYNC_TO_LDS_B128 + s_wait_asynccnt.
__global__ void __launch_bounds__(128) k_attn(
        const __bf16* __restrict__ Qbf, const __bf16* __restrict__ Kbf,
        const __bf16* __restrict__ Vt, __bf16* __restrict__ Ctx) {
    __shared__ __bf16 ks[2][32][40];           // [buf][key][d]   80B rows
    __shared__ __bf16 vs[2][32][40];           // [buf][d][key]   80B rows
    __shared__ __bf16 pbuf[4][16][40];         // per-wave P staging

    const int tid  = threadIdx.x;
    const int w    = tid >> 5, lane = tid & 31;
    const int half = lane >> 4, ln = lane & 15;
    const int bh   = blockIdx.y, b = bh >> 3, h = bh & 7;
    const int q0   = blockIdx.x * 64 + w * 16;
    const float scale = 0.17677669529663687f;  // 1/sqrt(32)

    const __bf16* Kbase = Kbf + (size_t)b * SEQ_L * D_MODEL + h * HEAD_D;
    const __bf16* Vbase = Vt + (size_t)b * D_MODEL * SEQ_L
                        + (size_t)(h * HEAD_D) * SEQ_L;

    // staging role: 128 threads cover 32 rows x 4 x 16B chunks
    const int srow = tid >> 2;          // 0..31
    const int sel  = (tid & 3) * 8;     // element offset 0,8,16,24

    // Q A-fragment: 16 rows x K=32 (= head dim) loaded once
    const __bf16* qp = Qbf + (size_t)(b * SEQ_L + q0 + ln) * D_MODEL
                     + h * HEAD_D + 8 * half;
    v16bf aq = load_frag(qp, 16);

    // prologue: stage tile 0
    async_ld16(&ks[0][srow][sel], Kbase + (size_t)srow * D_MODEL + sel);
    async_ld16(&vs[0][srow][sel], Vbase + (size_t)srow * SEQ_L + sel);

    float mrun[8], lrun[8];
    v8f acc0 = {}, acc1 = {};
    #pragma unroll
    for (int r = 0; r < 8; ++r) { mrun[r] = -3.0e38f; lrun[r] = 0.f; }

    const int nIt = SEQ_L / 32;
    for (int it = 0; it < nIt; ++it) {
        const int cur = it & 1;
        if (it + 1 < nIt) {                     // prefetch next 32-key slice
            const int ktn = (it + 1) * 32;
            async_ld16(&ks[cur ^ 1][srow][sel],
                       Kbase + (size_t)(ktn + srow) * D_MODEL + sel);
            async_ld16(&vs[cur ^ 1][srow][sel],
                       Vbase + (size_t)srow * SEQ_L + ktn + sel);
            S_WAIT_ASYNCCNT(2);                 // in-order: current slice done
        } else {
            S_WAIT_ASYNCCNT(0);
        }
        __syncthreads();

        // scores: S = Q * K^T  (two 16-key tiles, one WMMA each)
        v16bf bk0 = load_frag(&ks[cur][ln][8 * half], 16);
        v16bf bk1 = load_frag(&ks[cur][16 + ln][8 * half], 16);
        v8f z0 = {}, z1 = {};
        v8f s0 = wmma_bf16(aq, bk0, z0);
        v8f s1 = wmma_bf16(aq, bk1, z1);

        // online softmax: rows live in 16-lane groups (C-matrix layout)
        float p0[8], p1[8];
        #pragma unroll
        for (int r = 0; r < 8; ++r) {
            float a0 = s0[r] * scale, a1 = s1[r] * scale;
            float mx = fmaxf(a0, a1);
            #pragma unroll
            for (int off = 8; off; off >>= 1)
                mx = fmaxf(mx, __shfl_xor(mx, off, 32));
            float mnew = fmaxf(mrun[r], mx);
            float fac  = __expf(mrun[r] - mnew);
            float e0 = __expf(a0 - mnew), e1 = __expf(a1 - mnew);
            float sum = e0 + e1;
            #pragma unroll
            for (int off = 8; off; off >>= 1)
                sum += __shfl_xor(sum, off, 32);
            lrun[r] = lrun[r] * fac + sum;
            mrun[r] = mnew;
            acc0[r] *= fac; acc1[r] *= fac;
            p0[r] = e0; p1[r] = e1;
        }

        // P: C-layout -> A-layout via per-wave LDS (compiler inserts ds waits)
        #pragma unroll
        for (int r = 0; r < 8; ++r) {
            int row = 8 * half + r;
            pbuf[w][row][ln]      = (__bf16)p0[r];
            pbuf[w][row][16 + ln] = (__bf16)p1[r];
        }
        v16bf ap = load_frag(&pbuf[w][ln][8 * half], 16);

        // ctx += P * V  (V slice transposed in LDS -> contiguous B-fragments)
        v16bf bv0 = load_frag(&vs[cur][ln][8 * half], 16);
        v16bf bv1 = load_frag(&vs[cur][16 + ln][8 * half], 16);
        acc0 = wmma_bf16(ap, bv0, acc0);
        acc1 = wmma_bf16(ap, bv1, acc1);

        __syncthreads();                        // slice free before overwrite
    }

    #pragma unroll
    for (int r = 0; r < 8; ++r) {
        float inv = 1.0f / lrun[r];
        int m = q0 + 8 * half + r;
        size_t base = (size_t)(b * SEQ_L + m) * D_MODEL + h * HEAD_D;
        Ctx[base + ln]      = (__bf16)(acc0[r] * inv);
        Ctx[base + 16 + ln] = (__bf16)(acc1[r] * inv);
    }
}

// ---------------- Kernel 3: output projection + residual ----------------
__global__ void __launch_bounds__(256) k_oproj(
        const __bf16* __restrict__ Ctx, const __bf16* __restrict__ Wbf,
        const float* __restrict__ bo, const float* __restrict__ q_in,
        float* __restrict__ Proj) {
    __shared__ __bf16 xs[64][264];
    const int t0 = blockIdx.x * 64;
    const int tid = threadIdx.x, w = tid >> 5, lane = tid & 31;
    const int half = lane >> 4, ln = lane & 15;

    __builtin_prefetch(q_in + (size_t)t0 * D_MODEL, 0, 1);  // residual rows

    // stage bf16 ctx tile via async global->LDS copies (8 per thread)
    for (int idx = tid * 8; idx < 64 * D_MODEL; idx += 256 * 8) {
        int r = idx >> 8, c = idx & 255;
        async_ld16(&xs[r][c], Ctx + (size_t)t0 * D_MODEL + idx);
    }
    S_WAIT_ASYNCCNT(0);
    __syncthreads();

    const __bf16* W = Wbf + 3 * D_MODEL * D_MODEL;      // Wo
    for (int t = w; t < 64; t += 8) {
        int mt = t >> 4, nt = t & 15;
        v8f acc = {};
        #pragma unroll
        for (int kk = 0; kk < D_MODEL; kk += 32) {
            v16bf a = load_frag(&xs[mt * 16 + ln][kk + 8 * half], 16);
            v16bf bm = load_frag(W + (size_t)(nt * 16 + ln) * D_MODEL
                                   + kk + 8 * half, 16);
            acc = wmma_bf16(a, bm, acc);
        }
        int n = nt * 16 + ln;
        float bb = bo[n];
        #pragma unroll
        for (int r = 0; r < 8; ++r) {
            size_t row = (size_t)t0 + mt * 16 + 8 * half + r;
            Proj[row * D_MODEL + n] = acc[r] + bb + q_in[row * D_MODEL + n];
        }
    }
}

// ---------------- Kernel 4: LayerNorm ----------------
__global__ void __launch_bounds__(256) k_ln(
        const float* __restrict__ Proj, const float* __restrict__ g,
        const float* __restrict__ be, float* __restrict__ out) {
    __shared__ float red[16];
    const size_t row = blockIdx.x;
    const int tid = threadIdx.x;
    float x = Proj[row * D_MODEL + tid];
    float s = x, s2 = x * x;
    #pragma unroll
    for (int off = 16; off; off >>= 1) {
        s  += __shfl_xor(s,  off, 32);
        s2 += __shfl_xor(s2, off, 32);
    }
    if ((tid & 31) == 0) { red[tid >> 5] = s; red[8 + (tid >> 5)] = s2; }
    __syncthreads();
    s = 0.f; s2 = 0.f;
    #pragma unroll
    for (int i = 0; i < 8; ++i) { s += red[i]; s2 += red[8 + i]; }
    float mu  = s * (1.0f / 256.0f);
    float var = s2 * (1.0f / 256.0f) - mu * mu;
    float rs  = rsqrtf(var + 1e-5f);
    out[row * D_MODEL + tid] = (x - mu) * rs * g[tid] + be[tid];
}

// ---------------- host launcher ----------------
extern "C" void kernel_launch(void* const* d_in, const int* in_sizes, int n_in,
                              void* d_out, int out_size, void* d_ws, size_t ws_size,
                              hipStream_t stream) {
    const float* query = (const float*)d_in[0];
    const float* key_  = (const float*)d_in[1];
    const float* value = (const float*)d_in[2];
    const float* Wq = (const float*)d_in[3];
    const float* bq = (const float*)d_in[4];
    const float* Wk = (const float*)d_in[5];
    const float* bk = (const float*)d_in[6];
    const float* Wv = (const float*)d_in[7];
    const float* bv = (const float*)d_in[8];
    const float* Wo = (const float*)d_in[9];
    const float* bo = (const float*)d_in[10];
    const float* ln_g = (const float*)d_in[11];
    const float* ln_b = (const float*)d_in[12];
    float* out = (float*)d_out;

    char* ws = (char*)d_ws;
    const size_t wbytes = (size_t)4 * D_MODEL * D_MODEL * 2;     // 512 KB
    const size_t tbytes = (size_t)N_B * SEQ_L * D_MODEL * 2;     // 4 MB
    __bf16* Wbf = (__bf16*)(ws);
    __bf16* Qbf = (__bf16*)(ws + wbytes);
    __bf16* Kbf = (__bf16*)(ws + wbytes + tbytes);
    __bf16* Vt  = (__bf16*)(ws + wbytes + 2 * tbytes);
    __bf16* Ctx = (__bf16*)(ws + wbytes + 3 * tbytes);
    float*  Prj = (float*) (ws + wbytes + 4 * tbytes);

    k_convert_w<<<256, 256, 0, stream>>>(Wq, Wk, Wv, Wo, Wbf);
    k_qkv<<<(N_B * SEQ_L) / 64, 256, 0, stream>>>(query, key_, value, Wbf,
                                                  bq, bk, bv, Qbf, Kbf, Vt);
    k_attn<<<dim3(SEQ_L / 64, N_B * N_H), 128, 0, stream>>>(Qbf, Kbf, Vt, Ctx);
    k_oproj<<<(N_B * SEQ_L) / 64, 256, 0, stream>>>(Ctx, Wbf, bo, query, Prj);
    k_ln<<<N_B * SEQ_L, 256, 0, stream>>>(Prj, ln_g, ln_b, out);
}
